// NerfRender_30709016166577
// MI455X (gfx1250) — matrix operand order repaired and checked by reading, not compiled, not run on hardware
//
#include <hip/hip_runtime.h>
#include <hip/hip_bf16.h>

typedef __attribute__((ext_vector_type(16))) _Float16 v16h;
typedef __attribute__((ext_vector_type(8)))  _Float16 v8h;
typedef __attribute__((ext_vector_type(8)))  float    v8f;
typedef __attribute__((ext_vector_type(4)))  unsigned u32x4;
typedef __attribute__((ext_vector_type(8)))  int      i32x8;
typedef __attribute__((ext_vector_type(4)))  int      i32x4;

#define WIDTH 256
#define NEARP 0.1f
#define FARP  4.0f

// LDS layout (bytes): double-buffered weight K-half tiles + activations
#define LDS_WT0_OFF  0        // 64 KB: weight buffer 0 (stages 0,2,4,...)
#define LDS_WT1_OFF  65536    // 64 KB: weight buffer 1 (stages 1,3,5,...)
#define LDS_ACT_OFF  131072   // 8 waves * 16*256 f16 = 65536 B
#define LDS_DEP_OFF  196608   // 128 * f32 = 512 B
#define LDS_RGBD_OFF 197120   // 128 * 4 * f32 = 2048 B
#define LDS_TOTAL    199168

// ---------------------------------------------------------------------------
// Prep: convert/transpose weights to f16 once.
//   wt0[n][k] (k in [0,64), rows 60..63 zero-padded) = w_in[k][n]
//   wth[l][n][k] = w_hid[l][k][n]
// ---------------------------------------------------------------------------
__global__ void nerf_prep_weights(const float* __restrict__ w_in,
                                  const float* __restrict__ w_hid,
                                  _Float16* __restrict__ wt0,
                                  _Float16* __restrict__ wth) {
  int e = blockIdx.x * blockDim.x + threadIdx.x;
  if (e < 16384) {
    int n = e >> 6, k = e & 63;
    float v = (k < 60) ? w_in[k * WIDTH + n] : 0.0f;
    wt0[e] = (_Float16)v;  // e == n*64 + k
  }
  int e2 = e - 16384;
  if (e2 >= 0 && e2 < 7 * 65536) {
    int l = e2 >> 16;
    int rem = e2 & 65535;
    int n = rem >> 8, k = rem & 255;
    wth[e2] = (_Float16)w_hid[l * 65536 + k * WIDTH + n];
  }
}

// ---------------------------------------------------------------------------
// TDM: DMA a 2D f16 tile (tile0 elems/row x tile1 rows, row stride `stride0`
// elems in global memory, rows packed contiguously in LDS) into LDS.
// Descriptor per CDNA5 ISA ch.8 (D# groups); 2D -> groups 2/3 zero.
// ---------------------------------------------------------------------------
__device__ __forceinline__ void tdm_load_tile(const void* gsrc, unsigned lds_off,
                                              int tile0, int tile1, int stride0) {
  unsigned long long ga = (unsigned long long)(uintptr_t)gsrc;
  u32x4 g0;
  g0[0] = 1u;                                            // count=1, user D#
  g0[1] = lds_off;                                       // lds_addr (bytes)
  g0[2] = (unsigned)ga;                                  // global_addr[31:0]
  g0[3] = (unsigned)((ga >> 32) & 0x01FFFFFFull)         // global_addr[56:32]
          | (2u << 30);                                  // type = 2 ("image")
  i32x8 g1;
  g1[0] = (1 << 16);                                     // data_size = 2 bytes
  g1[1] = (tile0 & 0xFFFF) << 16;                        // tensor_dim0 = tile0
  g1[2] = (tile1 & 0xFFFF) << 16;                        // tensor_dim1 = tile1
  g1[3] = (tile0 & 0xFFFF) << 16;                        // tile_dim0
  g1[4] = (tile1 & 0xFFFF);                              // tile_dim1 (tile_dim2=0)
  g1[5] = stride0;                                       // tensor_dim0_stride lo
  g1[6] = 0;                                             // stride hi / dim1_stride
  g1[7] = 0;
  i32x4 z4 = {0, 0, 0, 0};
#if defined(__clang_major__) && (__clang_major__ >= 23)
  i32x8 z8 = {0, 0, 0, 0, 0, 0, 0, 0};
  __builtin_amdgcn_tensor_load_to_lds(g0, g1, z4, z4, z8, 0);
#else
  __builtin_amdgcn_tensor_load_to_lds(g0, g1, z4, z4, 0);
#endif
}

// ---------------------------------------------------------------------------
// Accumulate one K-range of a dense layer for one wave's 16-point tile.
//   actW: [16][256] f16 wave-private LDS (A operand, global k = kbase+kt*32)
//   Wt:   [256][strideK] f16 tile (k-contiguous per output column)
// ---------------------------------------------------------------------------
__device__ __forceinline__ void mlp_half_wmma(const _Float16* __restrict__ actW,
                                              const _Float16* __restrict__ Wt,
                                              int strideK, int KT, int kbase,
                                              int lane, v8f (&acc)[16]) {
  const int nl = lane & 15;
  const int hi = (lane >> 4) & 1;
  for (int kt = 0; kt < KT; ++kt) {
    const int kg = kbase + kt * 32;
    // A fragment (16x32, contiguous k order, same as B): row m = nl;
    // lanes 0-15: K {kg+0..15}; lanes 16-31: K {kg+16..31}
    v8h alo = *(const v8h*)(actW + nl * WIDTH + kg + hi * 16);
    v8h ahi = *(const v8h*)(actW + nl * WIDTH + kg + hi * 16 + 8);
    v16h a;
#pragma unroll
    for (int j = 0; j < 8; ++j) { a[j] = alo[j]; a[j + 8] = ahi[j]; }
#pragma unroll
    for (int nt = 0; nt < 16; ++nt) {
      // B fragment: column n = nt*16+nl; halves = local K {kt*32+hi*16 .. +15}
      v16h b = *(const v16h*)(Wt + (nt * 16 + nl) * strideK + kt * 32 + hi * 16);
      acc[nt] = __builtin_amdgcn_wmma_f32_16x16x32_f16(
          false, a, false, b, (short)0, acc[nt], false, false);
    }
  }
}

__device__ __forceinline__ void store_act_relu(_Float16* __restrict__ actW,
                                               const float* __restrict__ bias,
                                               int lane, v8f (&acc)[16]) {
  const int nl = lane & 15;
  const int hi = (lane >> 4) & 1;
#pragma unroll
  for (int nt = 0; nt < 16; ++nt) {
    const int n = nt * 16 + nl;
    const float bv = bias[n];
#pragma unroll
    for (int i = 0; i < 8; ++i) {
      float v = acc[nt][i] + bv;
      v = v > 0.0f ? v : 0.0f;
      actW[(hi * 8 + i) * WIDTH + n] = (_Float16)v;
    }
  }
}

// ---------------------------------------------------------------------------
// Main fused kernel: block = 128 points = 2 rays; 8 waves x 16 points each.
// Weight tiles TDM-prefetched (double-buffered) while WMMAs run.
// ---------------------------------------------------------------------------
__global__ __launch_bounds__(256) void nerf_forward_kernel(
    const float* __restrict__ ray_pos, const float* __restrict__ ray_dir,
    const float* __restrict__ jitter,
    const float* __restrict__ b_in,  const float* __restrict__ b_hid,
    const float* __restrict__ w_rgb, const float* __restrict__ b_rgb,
    const float* __restrict__ w_den, const float* __restrict__ b_den,
    const _Float16* __restrict__ wt0, const _Float16* __restrict__ wth,
    float* __restrict__ out) {
  extern __shared__ __align__(32) char smem[];
  _Float16* WtB[2] = { (_Float16*)(smem + LDS_WT0_OFF),
                       (_Float16*)(smem + LDS_WT1_OFF) };
  _Float16* act     = (_Float16*)(smem + LDS_ACT_OFF);
  float*    depthsS = (float*)(smem + LDS_DEP_OFF);
  float*    rgbdenS = (float*)(smem + LDS_RGBD_OFF);
  // LDS aperture: low 32 bits of the generic pointer == LDS byte offset
  const unsigned wtOff[2] = { (unsigned)(uintptr_t)WtB[0],
                              (unsigned)(uintptr_t)WtB[1] };

  const int tid  = threadIdx.x;
  const int wave = tid >> 5;
  const int lane = tid & 31;
  const int p0   = blockIdx.x * 128;
  _Float16* actW = act + wave * 16 * WIDTH;

  // Kick off layer-0 weight DMA (256x64 f16 = 32 KB) before encoding.
  if (tid == 0) tdm_load_tile(wt0, wtOff[0], 64, 256, 64);

  // ---- Fourier encoding into wave's activation tile (cols 60..63 zero)
  for (int idx = lane; idx < 16 * 64; idx += 32) {
    const int mrow = idx >> 6;
    const int f    = idx & 63;
    const int pl   = wave * 16 + mrow;
    const int p    = p0 + pl;
    const int ray  = p >> 6;
    const int s    = p & 63;
    const float depth =
        NEARP + (FARP - NEARP) * ((float)s + jitter[p]) * (1.0f / 64.0f);
    if (f == 0) depthsS[pl] = depth;
    float val = 0.0f;
    if (f < 60) {
      const int c   = f / 20;
      const int rem = f - c * 20;
      const int bb  = (rem < 10) ? rem : (rem - 10);
      const float x  = ray_pos[ray * 3 + c] + depth * ray_dir[ray * 3 + c];
      const float xb = x * (float)(1 << bb);
      val = (rem < 10) ? sinf(xb) : cosf(xb);
    }
    actW[mrow * WIDTH + f] = (_Float16)val;
  }

  if (tid == 0) __builtin_amdgcn_s_wait_tensorcnt(0);
  __syncthreads();  // layer-0 weights resident

  // ---- Layer 0 (K=64, 2 K-chunks) with stage-1 prefetch overlapped
  if (tid == 0) tdm_load_tile(wth, wtOff[1], 128, 256, 256);  // stage 1
  {
    v8f acc[16];
#pragma unroll
    for (int nt = 0; nt < 16; ++nt) { v8f z = {}; acc[nt] = z; }
    mlp_half_wmma(actW, WtB[0], 64, 2, 0, lane, acc);
    store_act_relu(actW, b_in, lane, acc);
  }

  // ---- 7 hidden layers; stages 1..14 = (layer, K-half) 64 KB tiles
  for (int l = 0; l < 7; ++l) {
    v8f acc[16];
#pragma unroll
    for (int nt = 0; nt < 16; ++nt) { v8f z = {}; acc[nt] = z; }
#pragma unroll
    for (int h = 0; h < 2; ++h) {
      const int s = 1 + l * 2 + h;
      if (tid == 0) __builtin_amdgcn_s_wait_tensorcnt(0);  // stage s arrived
      __syncthreads();
      const int nx = s + 1;  // prefetch next stage into the other buffer
      if (tid == 0 && nx <= 14) {
        const int nl_ = (nx - 1) >> 1, nh = (nx - 1) & 1;
        tdm_load_tile(wth + nl_ * 65536 + nh * 128, wtOff[nx & 1],
                      128, 256, 256);
      }
      mlp_half_wmma(actW, WtB[s & 1], 128, 4, h * 128, lane, acc);
      __syncthreads();  // all waves done reading buffer s&1
    }
    store_act_relu(actW, b_hid + l * WIDTH, lane, acc);
  }

  // ---- Heads: rgb = sigmoid(act@w_rgb + b), density = relu(act@w_den + b)
  {
    const int m  = lane & 15;
    const int hi = lane >> 4;
    const _Float16* arow = actW + m * WIDTH + hi * 128;
    float r = 0.0f, g = 0.0f, b2 = 0.0f, dsum = 0.0f;
    for (int k = 0; k < 128; ++k) {
      const float a  = (float)arow[k];
      const int   kk = hi * 128 + k;
      r    += a * w_rgb[kk * 3 + 0];
      g    += a * w_rgb[kk * 3 + 1];
      b2   += a * w_rgb[kk * 3 + 2];
      dsum += a * w_den[kk];
    }
    r    += __shfl_xor(r, 16);
    g    += __shfl_xor(g, 16);
    b2   += __shfl_xor(b2, 16);
    dsum += __shfl_xor(dsum, 16);
    if (hi == 0) {
      const int pl = wave * 16 + m;
      rgbdenS[pl * 4 + 0] = 1.0f / (1.0f + expf(-(r  + b_rgb[0])));
      rgbdenS[pl * 4 + 1] = 1.0f / (1.0f + expf(-(g  + b_rgb[1])));
      rgbdenS[pl * 4 + 2] = 1.0f / (1.0f + expf(-(b2 + b_rgb[2])));
      rgbdenS[pl * 4 + 3] = fmaxf(dsum + b_den[0], 0.0f);
    }
  }
  __syncthreads();

  // ---- Volume rendering: 2 rays per block, sequential compositing
  if (tid < 2) {
    const int q = tid;
    const int r = blockIdx.x * 2 + q;
    float T = 1.0f, cr = 0.0f, cg = 0.0f, cb = 0.0f, cd = 0.0f;
    for (int s = 0; s < 64; ++s) {
      const float dep   = depthsS[q * 64 + s];
      const float delta = (s < 63) ? (depthsS[q * 64 + s + 1] - dep) : 1e10f;
      const float den   = rgbdenS[(q * 64 + s) * 4 + 3];
      const float tau   = den * delta;
      const float e     = expf(-tau);
      const float w     = (1.0f - e) * T;
      cr += w * rgbdenS[(q * 64 + s) * 4 + 0];
      cg += w * rgbdenS[(q * 64 + s) * 4 + 1];
      cb += w * rgbdenS[(q * 64 + s) * 4 + 2];
      cd += w * dep;
      T *= e;
    }
    out[r * 4 + 0] = cr; out[r * 4 + 1] = cg;
    out[r * 4 + 2] = cb; out[r * 4 + 3] = cd;
  }
}

// ---------------------------------------------------------------------------
extern "C" void kernel_launch(void* const* d_in, const int* in_sizes, int n_in,
                              void* d_out, int out_size, void* d_ws, size_t ws_size,
                              hipStream_t stream) {
  (void)in_sizes; (void)n_in; (void)out_size; (void)ws_size;
  const float* ray_pos = (const float*)d_in[0];
  const float* ray_dir = (const float*)d_in[1];
  const float* jitter  = (const float*)d_in[2];
  const float* w_in    = (const float*)d_in[3];
  const float* b_in    = (const float*)d_in[4];
  const float* w_hid   = (const float*)d_in[5];
  const float* b_hid   = (const float*)d_in[6];
  const float* w_rgb   = (const float*)d_in[7];
  const float* b_rgb   = (const float*)d_in[8];
  const float* w_den   = (const float*)d_in[9];
  const float* b_den   = (const float*)d_in[10];
  float* out = (float*)d_out;

  _Float16* wt0 = (_Float16*)d_ws;   // 16384 halves
  _Float16* wth = wt0 + 16384;       // 7*65536 halves

  nerf_prep_weights<<<1856, 256, 0, stream>>>(w_in, w_hid, wt0, wth);

  hipFuncSetAttribute((const void*)nerf_forward_kernel,
                      hipFuncAttributeMaxDynamicSharedMemorySize, LDS_TOTAL);
  nerf_forward_kernel<<<2048, 256, LDS_TOTAL, stream>>>(
      ray_pos, ray_dir, jitter, b_in, b_hid, w_rgb, b_rgb, w_den, b_den,
      wt0, wth, out);
}
